// Attention_38027640439268
// MI455X (gfx1250) — matrix operand order repaired and checked by reading, not compiled, not run on hardware
//
#include <hip/hip_runtime.h>
#include <hip/hip_bf16.h>
#include <math.h>

typedef __bf16 bf16_t;
typedef bf16_t v16bf __attribute__((ext_vector_type(16)));
typedef bf16_t v8bf  __attribute__((ext_vector_type(8)));
typedef float  v8f   __attribute__((ext_vector_type(8)));
typedef unsigned int v4u __attribute__((ext_vector_type(4)));
typedef int v8i __attribute__((ext_vector_type(8)));
typedef int v4i __attribute__((ext_vector_type(4)));

#define B_   2
#define S_   2048
#define D_   1024
#define NH_  16
#define H_   64
#define IGNORE_ (-100000.0f)

static __device__ __forceinline__ v8f wmma_bf16(v16bf a, v16bf b, v8f c) {
  return __builtin_amdgcn_wmma_f32_16x16x32_bf16(false, a, false, b, (short)0, c,
                                                 false, false);
}

// A-operand (16x32, 16-bit, row-major, leading dim ld):
// lanes 0-15: row m=lane, K = k0+{0..7,16..23}; lanes 16-31: row m=lane-16,
// K = k0+{8..15,24..31}.  Two 16B loads per lane.
static __device__ __forceinline__ v16bf load_fragA(const bf16_t* p, int ld,
                                                   int row0, int k0) {
  int lane = threadIdx.x & 31;
  int m  = lane & 15;
  int hb = (lane >> 4) << 3;
  const bf16_t* base = p + (size_t)(row0 + m) * ld + k0 + hb;
  v8bf lo = *(const v8bf*)(base);
  v8bf hi = *(const v8bf*)(base + 16);
  v16bf r;
#pragma unroll
  for (int i = 0; i < 8; ++i) { r[i] = lo[i]; r[8 + i] = hi[i]; }
  return r;
}

// B-operand (32x16, 16-bit), column n contiguous in K: element (k,n) at
// p[(n0+n)*ld + k]. Lanes 0-15: K=k0+{0..7,16..23} of col lane; lanes 16-31:
// K=k0+{8..15,24..31} of col lane-16.  Two 16B loads per lane.
static __device__ __forceinline__ v16bf load_fragB(const bf16_t* p, int ld,
                                                   int n0, int k0) {
  int lane = threadIdx.x & 31;
  int n  = lane & 15;
  int kh = (lane >> 4) << 3;
  const bf16_t* base = p + (size_t)(n0 + n) * ld + k0 + kh;
  v8bf lo = *(const v8bf*)(base);
  v8bf hi = *(const v8bf*)(base + 16);
  v16bf r;
#pragma unroll
  for (int i = 0; i < 8; ++i) { r[i] = lo[i]; r[8 + i] = hi[i]; }
  return r;
}

// 1-D Tensor Data Mover copy: contiguous `bytes` (multiple of 8, <=512KB)
// from global to LDS.  D# per ISA cdna5 ch.8: count=1, type=2(image),
// data_size=3 (8B units), 2-D degenerate tensor (dim1=1).
// 6-arg builtin form: (g0:v4u, g1:v8i, g2:v4i, g3:v4i, g4:v8i, cpol imm).
static __device__ __forceinline__ void tdm_load_lds(bf16_t* lds_dst,
                                                    const bf16_t* gsrc,
                                                    unsigned bytes) {
  unsigned units = bytes >> 3;                       // 8-byte elements
  unsigned lds   = (unsigned)(uintptr_t)lds_dst;     // LDS addr = addr[31:0]
  unsigned long long ga = (unsigned long long)(uintptr_t)gsrc;
  v4u g0;
  g0[0] = 1u;                                        // count=1, user mode
  g0[1] = lds;                                       // lds_addr [63:32]
  g0[2] = (unsigned)ga;                              // global_addr lo
  g0[3] = (unsigned)((ga >> 32) & 0x01FFFFFFu) | (2u << 30);  // addr hi | type=2
  v8i g1;
  g1[0] = (int)(3u << 16);                           // data_size = 8B
  g1[1] = (int)((units & 0xFFFFu) << 16);            // tensor_dim0 lo16
  g1[2] = (int)((units >> 16) & 0xFFFFu) | (1 << 16);// dim0 hi16 | tensor_dim1=1
  g1[3] = (int)(units << 16);                        // tile_dim0
  g1[4] = 1;                                         // tile_dim1 = 1
  g1[5] = (int)units;                                // tensor_dim0_stride lo32
  g1[6] = 0;
  g1[7] = 0;
  v4i z4 = {0, 0, 0, 0};
  v8i z8 = {0, 0, 0, 0, 0, 0, 0, 0};
  __builtin_amdgcn_tensor_load_to_lds(g0, g1, z4, z4, z8, 0);
}

// ---------------------------------------------------------------------------
// Prep kernel A: x (f32) -> Xbf (bf16), contiguous, vectorized.
// ---------------------------------------------------------------------------
__global__ __launch_bounds__(256) void xconv_kernel(const float* __restrict__ x,
                                                    bf16_t* __restrict__ Xbf) {
  size_t i = ((size_t)blockIdx.x * 256 + threadIdx.x) * 8;
  float4 f0 = *(const float4*)(x + i);
  float4 f1 = *(const float4*)(x + i + 4);
  v8bf o;
  o[0] = (bf16_t)f0.x; o[1] = (bf16_t)f0.y; o[2] = (bf16_t)f0.z; o[3] = (bf16_t)f0.w;
  o[4] = (bf16_t)f1.x; o[5] = (bf16_t)f1.y; o[6] = (bf16_t)f1.z; o[7] = (bf16_t)f1.w;
  *(v8bf*)(Xbf + i) = o;
}

// ---------------------------------------------------------------------------
// Prep kernel B: weights -> bf16, transposed so WMMA-K is contiguous.
// W_{Q,K,V} (n,d,h) -> WT[n][h][d];  W_O (nh,d) -> WoT[d][nh].
// ---------------------------------------------------------------------------
__global__ __launch_bounds__(256) void wprep_kernel(
    const float* __restrict__ Wq, const float* __restrict__ Wk,
    const float* __restrict__ Wv, const float* __restrict__ Wo,
    bf16_t* __restrict__ WqT, bf16_t* __restrict__ WkT,
    bf16_t* __restrict__ WvT, bf16_t* __restrict__ WoT) {
  unsigned o = blockIdx.x * 256 + threadIdx.x;       // 4M total
  unsigned mat = o >> 20;
  unsigned r   = o & 0xFFFFFu;
  if (mat < 3) {
    const float* W = (mat == 0) ? Wq : (mat == 1) ? Wk : Wv;
    bf16_t* T      = (mat == 0) ? WqT : (mat == 1) ? WkT : WvT;
    unsigned nh = r >> 10, d = r & 1023;
    unsigned n = nh >> 6, h = nh & 63;
    T[r] = (bf16_t)W[((size_t)n * D_ + d) * H_ + h];
  } else {
    unsigned d = r >> 10, nh = r & 1023;
    WoT[r] = (bf16_t)Wo[(size_t)nh * D_ + d];
  }
}

// ---------------------------------------------------------------------------
// Kernel 1: QKV projection, 32 rows x 16 cols per wave.
// TDM stages a contiguous 64KB bf16 X tile into LDS.  Q scaled by 1/8,
// V stored transposed [b][n][h][s].
// ---------------------------------------------------------------------------
__global__ __launch_bounds__(256) void qkv_kernel(
    const bf16_t* __restrict__ Xbf,
    const bf16_t* __restrict__ WqT, const bf16_t* __restrict__ WkT,
    const bf16_t* __restrict__ WvT,
    const float* __restrict__ bq, const float* __restrict__ bk,
    const float* __restrict__ bv,
    bf16_t* __restrict__ Q, bf16_t* __restrict__ K, bf16_t* __restrict__ Vt) {
  __shared__ bf16_t xt[32 * D_];                     // 64KB
  int tid  = threadIdx.x;
  int row0 = blockIdx.x * 32;
  if (tid == 0) {
    tdm_load_lds(xt, Xbf + (size_t)row0 * D_, 32 * D_ * 2);
    __builtin_amdgcn_s_wait_tensorcnt(0);
  }
  __syncthreads();

  int wave = tid >> 5, lane = tid & 31;
  int ct   = blockIdx.y * 8 + wave;                  // 0..191
  int mat  = ct >> 6;                                // 0=Q 1=K 2=V
  int head = (ct & 63) >> 2;
  int h0   = (ct & 3) << 4;
  const bf16_t* WT   = (mat == 0) ? WqT : (mat == 1) ? WkT : WvT;
  const float*  bias = (mat == 0) ? bq : (mat == 1) ? bk : bv;
  const bf16_t* WhT  = WT + (size_t)head * H_ * D_;  // [h][d]

  v8f acc[2] = {};
#pragma unroll 4
  for (int k0 = 0; k0 < D_; k0 += 32) {
    v16bf b  = load_fragB(WhT, D_, h0, k0);
    v16bf a0 = load_fragA(xt, D_, 0, k0);
    acc[0] = wmma_bf16(a0, b, acc[0]);
    v16bf a1 = load_fragA(xt, D_, 16, k0);
    acc[1] = wmma_bf16(a1, b, acc[1]);
  }

  float bcol  = bias[head * H_ + h0 + (lane & 15)];
  float scale = (mat == 0) ? 0.125f : 1.0f;
#pragma unroll
  for (int hf = 0; hf < 2; ++hf) {
#pragma unroll
    for (int r = 0; r < 8; ++r) {
      int row = row0 + hf * 16 + r + ((lane >> 4) << 3);
      int bb  = row >> 11;
      int s   = row & (S_ - 1);
      int h   = h0 + (lane & 15);
      float v = (acc[hf][r] + bcol) * scale;
      if (mat == 2)
        Vt[(((size_t)bb * NH_ + head) * H_ + h) * S_ + s] = (bf16_t)v;
      else {
        bf16_t* dst = (mat == 0) ? Q : K;
        dst[(((size_t)bb * NH_ + head) * S_ + s) * H_ + h] = (bf16_t)v;
      }
    }
  }
}

// ---------------------------------------------------------------------------
// Kernel 2: causal flash attention. One wave = 16 query rows, key tiles of 32.
// ---------------------------------------------------------------------------
__global__ __launch_bounds__(256) void attn_kernel(
    const bf16_t* __restrict__ Q, const bf16_t* __restrict__ K,
    const bf16_t* __restrict__ Vt, bf16_t* __restrict__ Z) {
  __shared__ bf16_t pbuf[8][16 * 32];
  int tid = threadIdx.x, wave = tid >> 5, lane = tid & 31;
  int q0   = blockIdx.x * 128 + wave * 16;
  int head = blockIdx.y, b = blockIdx.z;
  const bf16_t* Qh = Q  + (((size_t)b * NH_ + head) * S_) * H_;
  const bf16_t* Kh = K  + (((size_t)b * NH_ + head) * S_) * H_;
  const bf16_t* Vh = Vt + (((size_t)b * NH_ + head) * H_) * S_;
  bf16_t* pb = &pbuf[wave][0];

  v16bf qa0 = load_fragA(Qh, H_, q0, 0);
  v16bf qa1 = load_fragA(Qh, H_, q0, 32);

  float mr[8], lr[8];
  v8f acc[4] = {};
#pragma unroll
  for (int r = 0; r < 8; ++r) { mr[r] = -1.0e30f; lr[r] = 0.0f; }

  int nkt = (q0 + 16 + 31) >> 5;
  for (int kt = 0; kt < nkt; ++kt) {
    int kb = kt << 5;
    if (kt + 1 < nkt) {                    // prefetch next K/V tile
      __builtin_prefetch(Kh + (size_t)(kb + 32 + lane) * H_, 0, 3);
      __builtin_prefetch(Vh + (size_t)lane * S_ + kb + 32, 0, 3);
      __builtin_prefetch(Vh + (size_t)(lane + 32) * S_ + kb + 32, 0, 3);
    }
    v8f s0 = {}, s1 = {};
    s0 = wmma_bf16(qa0, load_fragB(Kh, H_, kb,      0),  s0);
    s0 = wmma_bf16(qa1, load_fragB(Kh, H_, kb,      32), s0);
    s1 = wmma_bf16(qa0, load_fragB(Kh, H_, kb + 16, 0),  s1);
    s1 = wmma_bf16(qa1, load_fragB(Kh, H_, kb + 16, 32), s1);

    if (kt == nkt - 1) {
#pragma unroll
      for (int r = 0; r < 8; ++r) {
        int row  = q0 + r + ((lane >> 4) << 3);
        int col0 = kb + (lane & 15);
        if (col0      > row) s0[r] = IGNORE_;
        if (col0 + 16 > row) s1[r] = IGNORE_;
      }
    }

#pragma unroll
    for (int r = 0; r < 8; ++r) {
      float a0 = s0[r], a1 = s1[r];
      float mt = fmaxf(a0, a1);
      mt = fmaxf(mt, __shfl_xor(mt, 1, 32));
      mt = fmaxf(mt, __shfl_xor(mt, 2, 32));
      mt = fmaxf(mt, __shfl_xor(mt, 4, 32));
      mt = fmaxf(mt, __shfl_xor(mt, 8, 32));
      float mnew = fmaxf(mr[r], mt);
      float p0 = __expf(a0 - mnew);
      float p1 = __expf(a1 - mnew);
      float rs = p0 + p1;
      rs += __shfl_xor(rs, 1, 32);
      rs += __shfl_xor(rs, 2, 32);
      rs += __shfl_xor(rs, 4, 32);
      rs += __shfl_xor(rs, 8, 32);
      float alpha = __expf(mr[r] - mnew);
      lr[r] = lr[r] * alpha + rs;
      mr[r] = mnew;
#pragma unroll
      for (int t = 0; t < 4; ++t) acc[t][r] *= alpha;
      int rl = r + ((lane >> 4) << 3);
      pb[rl * 32 + (lane & 15)]      = (bf16_t)p0;
      pb[rl * 32 + 16 + (lane & 15)] = (bf16_t)p1;
    }
    asm volatile("s_wait_dscnt 0" ::: "memory");

    v16bf pa = load_fragA(pb, 32, 0, 0);
#pragma unroll
    for (int t = 0; t < 4; ++t)
      acc[t] = wmma_bf16(pa, load_fragB(Vh, S_, t * 16, kb), acc[t]);
  }

#pragma unroll
  for (int t = 0; t < 4; ++t) {
    int h = t * 16 + (lane & 15);
#pragma unroll
    for (int r = 0; r < 8; ++r) {
      int s = q0 + r + ((lane >> 4) << 3);
      Z[(((size_t)b * S_ + s) * NH_ + head) * H_ + h] =
          (bf16_t)(acc[t][r] / lr[r]);
    }
  }
}

// ---------------------------------------------------------------------------
// Kernel 3: output projection. Zflat[4096x1024]bf16 @ WoT + b_O -> f32 out.
// TDM stages the contiguous 64KB Z tile.
// ---------------------------------------------------------------------------
__global__ __launch_bounds__(256) void oproj_kernel(
    const bf16_t* __restrict__ Zf, const bf16_t* __restrict__ WoT,
    const float* __restrict__ bo, float* __restrict__ out) {
  __shared__ bf16_t zt[32 * D_];                     // 64KB
  int tid  = threadIdx.x;
  int row0 = blockIdx.x * 32;
  if (tid == 0) {
    tdm_load_lds(zt, Zf + (size_t)row0 * D_, 32 * D_ * 2);
    __builtin_amdgcn_s_wait_tensorcnt(0);
  }
  __syncthreads();

  int wave = tid >> 5, lane = tid & 31;
  int d0 = (blockIdx.y * 8 + wave) * 16;
  v8f acc[2] = {};
#pragma unroll 4
  for (int k0 = 0; k0 < D_; k0 += 32) {
    v16bf b  = load_fragB(WoT, D_, d0, k0);          // WoT[d][nh]
    v16bf a0 = load_fragA(zt, D_, 0, k0);
    acc[0] = wmma_bf16(a0, b, acc[0]);
    v16bf a1 = load_fragA(zt, D_, 16, k0);
    acc[1] = wmma_bf16(a1, b, acc[1]);
  }
  float bcol = bo[d0 + (lane & 15)];
#pragma unroll
  for (int hf = 0; hf < 2; ++hf) {
#pragma unroll
    for (int r = 0; r < 8; ++r) {
      int row = row0 + hf * 16 + r + ((lane >> 4) << 3);
      out[(size_t)row * D_ + d0 + (lane & 15)] = acc[hf][r] + bcol;
    }
  }
}

// ---------------------------------------------------------------------------
extern "C" void kernel_launch(void* const* d_in, const int* in_sizes, int n_in,
                              void* d_out, int out_size, void* d_ws,
                              size_t ws_size, hipStream_t stream) {
  (void)in_sizes; (void)n_in; (void)out_size; (void)ws_size;
  const float* x  = (const float*)d_in[0];
  const float* Wq = (const float*)d_in[1];
  const float* Wk = (const float*)d_in[2];
  const float* Wv = (const float*)d_in[3];
  const float* Wo = (const float*)d_in[4];
  const float* bq = (const float*)d_in[5];
  const float* bk = (const float*)d_in[6];
  const float* bv = (const float*)d_in[7];
  const float* bo = (const float*)d_in[8];
  float* out = (float*)d_out;

  // workspace layout (bf16 elements):
  // Q,K,Vt,Z: 4 x 4M (32 MiB) | Xbf: 4M (8 MiB) | WqT/WkT/WvT/WoT: 4 x 1M (8 MiB)
  size_t elems = (size_t)B_ * NH_ * S_ * H_;         // 4M
  bf16_t* Q   = (bf16_t*)d_ws;
  bf16_t* K   = Q + elems;
  bf16_t* Vt  = K + elems;
  bf16_t* Z   = Vt + elems;
  bf16_t* Xbf = Z + elems;
  bf16_t* WqT = Xbf + (size_t)B_ * S_ * D_;
  bf16_t* WkT = WqT + (size_t)NH_ * H_ * D_;
  bf16_t* WvT = WkT + (size_t)NH_ * H_ * D_;
  bf16_t* WoT = WvT + (size_t)NH_ * H_ * D_;

  xconv_kernel<<<dim3(((size_t)B_ * S_ * D_) / (8 * 256)), 256, 0, stream>>>(x, Xbf);
  wprep_kernel<<<dim3((4u << 20) / 256), 256, 0, stream>>>(
      Wq, Wk, Wv, Wo, WqT, WkT, WvT, WoT);
  qkv_kernel<<<dim3((B_ * S_) / 32, 24), 256, 0, stream>>>(
      Xbf, WqT, WkT, WvT, bq, bk, bv, Q, K, Vt);
  attn_kernel<<<dim3(S_ / 128, NH_, B_), 256, 0, stream>>>(Q, K, Vt, Z);
  oproj_kernel<<<dim3((B_ * S_) / 32, D_ / 128), 256, 0, stream>>>(Z, WoT, bo, out);
}
